// CrAKNLayer_30554397343953
// MI455X (gfx1250) — compile-verified, matrix operand order checked by reading, not run on hardware
//
#include <hip/hip_runtime.h>

#define N_NODES   20000
#define N_EDGES   320000
#define DIM       256
#define LDS_PITCH 264   // 256 + 8 bf16 pad -> row stride 528B (132 banks, 4/row):
                        // conflict-free 16B ds_load_b128 fragment reads

typedef __attribute__((ext_vector_type(16))) __bf16       v16bf;
typedef __attribute__((ext_vector_type(8)))  float        v8f;
typedef __attribute__((ext_vector_type(4)))  float        v4f;
typedef __attribute__((ext_vector_type(4)))  __bf16       v4bf;
typedef __attribute__((ext_vector_type(4)))  unsigned int v4u;

union BF16Frag { v16bf v; v4u u4[2]; };
union Pack8    { v4bf  b[2]; v4u u; };

// 8 fp32 -> 8 bf16 (RNE) packed in 16 bytes; lets clang pick native cvt
__device__ __forceinline__ v4u cvt8_bf16(v4f f0, v4f f1) {
    Pack8 p;
    p.b[0] = __builtin_convertvector(f0, v4bf);
    p.b[1] = __builtin_convertvector(f1, v4bf);
    return p.u;
}

// mish(x) = x * tanh(softplus(x)).
// With u = 1 + e^x:  tanh(ln u) = (u^2 - 1)/(u^2 + 1), so
// mish(x) = x * (u^2 - 1) / (u^2 + 1).
// Branchless: 1 v_exp_f32 + 1 v_rcp_f32 + ~5 VALU. Clamp at 30: u^2 ~ 1.1e26,
// ratio rounds to 1.0f -> exact large-x limit mish -> x; x -> -inf gives
// ratio -> e^x -> mish -> 0- correctly.
__device__ __forceinline__ float mish_f(float v) {
    float e  = __expf(fminf(v, 30.0f));
    float u  = 1.0f + e;
    float u2 = u * u;
    return v * (u2 - 1.0f) * __builtin_amdgcn_rcpf(u2 + 1.0f);
}

// --- prep: zero agg, convert the three weight matrices fp32 -> bf16 ---------
__global__ void __launch_bounds__(256) prep_kernel(
    const float* __restrict__ Wd, const float* __restrict__ We,
    const float* __restrict__ Wo,
    __bf16* __restrict__ wd, __bf16* __restrict__ we, __bf16* __restrict__ wo,
    float* __restrict__ agg)
{
    int idx = blockIdx.x * 256 + threadIdx.x;
    if (idx < N_NODES * DIM) agg[idx] = 0.0f;
    if (idx < DIM * DIM) {
        wd[idx] = (__bf16)Wd[idx];
        we[idx] = (__bf16)We[idx];
        wo[idx] = (__bf16)Wo[idx];
    }
}

// B frag: B[k,n] = W[n,k]; lane holds column n; bf16 weight row n contiguous.
// 16-bit B 32x16 striping: lane<16 holds K = kb..kb+7 & kb+16..kb+23 (kb=kk),
// lane>=16 shifted by 8 -> same kb = kk + 8*half formula as A.
__device__ __forceinline__ void load_b_frag(BF16Frag& b, const v4u* Wrow, int kb) {
    b.u4[0] = Wrow[(kb >> 3) + 0];   // K = kb    .. kb+7
    b.u4[1] = Wrow[(kb >> 3) + 2];   // K = kb+16 .. kb+23
}

// --- kernel 1: x = mish(node @ Wd^T + bd) -----------------------------------
// One block (16 waves) per 16-row tile; wave w computes column tile w.
__global__ void __launch_bounds__(512) node_mlp_kernel(
    const float* __restrict__ A, const __bf16* __restrict__ Wbf,
    const float* __restrict__ bias, float* __restrict__ out)
{
    __shared__ __attribute__((aligned(16))) unsigned short As[16 * LDS_PITCH];

    const int tid = threadIdx.x;
    const int tm  = blockIdx.x;

    // cooperative stage: 16x256 fp32 tile -> bf16 LDS, one conversion per element
    {
        int t = tid * 8;                 // 512 threads x 8 elems = 4096
        int r = t >> 8;                  // row 0..15
        int c = t & 255;                 // k, multiple of 8
        const v4f* p = (const v4f*)(A + ((size_t)tm * 16 + r) * DIM + c);
        *(v4u*)&As[r * LDS_PITCH + c] = cvt8_bf16(p[0], p[1]);
    }
    __syncthreads();

    const int lane = tid & 31;
    const int warp = tid >> 5;           // tn = 0..15
    const int half = lane >> 4;
    const int mrow = lane & 15;
    const int col  = warp * 16 + mrow;

    const v4u* Wrow = (const v4u*)(Wbf + (size_t)col * DIM);
    const unsigned short* Arow = &As[mrow * LDS_PITCH];

    v8f acc = {};
    #pragma unroll
    for (int kk = 0; kk < DIM; kk += 32) {
        const int kb = kk + half * 8;
        BF16Frag a, b;
        a.u4[0] = *(const v4u*)&Arow[kb];        // ds_load_b128
        a.u4[1] = *(const v4u*)&Arow[kb + 16];
        load_b_frag(b, Wrow, kb);
        acc = __builtin_amdgcn_wmma_f32_16x16x32_bf16(
            false, a.v, false, b.v, (short)0, acc, false, false);
    }

    const float bv = bias[col];
    #pragma unroll
    for (int v = 0; v < 8; ++v) {
        int gm = tm * 16 + v + half * 8;         // C/D: M = v + 8*half, N = col
        out[(size_t)gm * DIM + col] = mish_f(acc[v] + bv);
    }
}

// --- kernel 2 (fused): y = mish(edge @ We^T + be);
//     msg = relu(x[src] + y); agg[dst] += msg --------------------------------
__global__ void __launch_bounds__(512) edge_mlp_scatter_kernel(
    const float* __restrict__ E, const __bf16* __restrict__ Wbf,
    const float* __restrict__ bias, const float* __restrict__ x,
    const int* __restrict__ src, const int* __restrict__ dst,
    float* __restrict__ agg)
{
    __shared__ __attribute__((aligned(16))) unsigned short As[16 * LDS_PITCH];
    __shared__ int s_src[16];
    __shared__ int s_dst[16];

    const int tid = threadIdx.x;
    const int tm  = blockIdx.x;                  // edge tile 0..19999

    if (tid < 16)      s_src[tid]      = src[tm * 16 + tid];
    else if (tid < 32) s_dst[tid - 16] = dst[tm * 16 + (tid - 16)];

    {
        int t = tid * 8;
        int r = t >> 8;
        int c = t & 255;
        const v4f* p = (const v4f*)(E + ((size_t)tm * 16 + r) * DIM + c);
        *(v4u*)&As[r * LDS_PITCH + c] = cvt8_bf16(p[0], p[1]);
    }
    __syncthreads();

    const int lane = tid & 31;
    const int warp = tid >> 5;
    const int half = lane >> 4;
    const int mrow = lane & 15;
    const int col  = warp * 16 + mrow;

    const v4u* Wrow = (const v4u*)(Wbf + (size_t)col * DIM);
    const unsigned short* Arow = &As[mrow * LDS_PITCH];

    v8f acc = {};
    #pragma unroll
    for (int kk = 0; kk < DIM; kk += 32) {
        const int kb = kk + half * 8;
        BF16Frag a, b;
        a.u4[0] = *(const v4u*)&Arow[kb];
        a.u4[1] = *(const v4u*)&Arow[kb + 16];
        load_b_frag(b, Wrow, kb);
        acc = __builtin_amdgcn_wmma_f32_16x16x32_bf16(
            false, a.v, false, b.v, (short)0, acc, false, false);
    }

    const float bv = bias[col];
    #pragma unroll
    for (int v = 0; v < 8; ++v) {
        int m = v + half * 8;
        float y = mish_f(acc[v] + bv);
        int s = s_src[m];                        // LDS broadcast
        int d = s_dst[m];
        float msg = x[(size_t)s * DIM + col] + y;   // gather: L2-resident (20.5 MB)
        msg = fmaxf(msg, 0.0f);
        atomicAdd(&agg[(size_t)d * DIM + col], msg); // L2-resident f32 atomics
    }
}

// --- kernel 3: out = mish((x + agg) @ Wo^T + bo) ----------------------------
__global__ void __launch_bounds__(512) out_mlp_kernel(
    const float* __restrict__ X, const float* __restrict__ G,
    const __bf16* __restrict__ Wbf, const float* __restrict__ bias,
    float* __restrict__ out)
{
    __shared__ __attribute__((aligned(16))) unsigned short As[16 * LDS_PITCH];

    const int tid = threadIdx.x;
    const int tm  = blockIdx.x;

    {
        int t = tid * 8;
        int r = t >> 8;
        int c = t & 255;
        size_t off = ((size_t)tm * 16 + r) * DIM + c;
        const v4f* px = (const v4f*)(X + off);
        const v4f* pg = (const v4f*)(G + off);
        *(v4u*)&As[r * LDS_PITCH + c] = cvt8_bf16(px[0] + pg[0], px[1] + pg[1]);
    }
    __syncthreads();

    const int lane = tid & 31;
    const int warp = tid >> 5;
    const int half = lane >> 4;
    const int mrow = lane & 15;
    const int col  = warp * 16 + mrow;

    const v4u* Wrow = (const v4u*)(Wbf + (size_t)col * DIM);
    const unsigned short* Arow = &As[mrow * LDS_PITCH];

    v8f acc = {};
    #pragma unroll
    for (int kk = 0; kk < DIM; kk += 32) {
        const int kb = kk + half * 8;
        BF16Frag a, b;
        a.u4[0] = *(const v4u*)&Arow[kb];
        a.u4[1] = *(const v4u*)&Arow[kb + 16];
        load_b_frag(b, Wrow, kb);
        acc = __builtin_amdgcn_wmma_f32_16x16x32_bf16(
            false, a.v, false, b.v, (short)0, acc, false, false);
    }

    const float bv = bias[col];
    #pragma unroll
    for (int v = 0; v < 8; ++v) {
        int gm = tm * 16 + v + half * 8;
        out[(size_t)gm * DIM + col] = mish_f(acc[v] + bv);
    }
}

extern "C" void kernel_launch(void* const* d_in, const int* in_sizes, int n_in,
                              void* d_out, int out_size, void* d_ws, size_t ws_size,
                              hipStream_t stream) {
    const float* node = (const float*)d_in[0];
    const float* edge = (const float*)d_in[1];
    /* d_in[2] targets: unused by forward */
    const int*   src  = (const int*)d_in[3];
    const int*   dst  = (const int*)d_in[4];
    const float* Wd   = (const float*)d_in[5];
    const float* bd   = (const float*)d_in[6];
    const float* We   = (const float*)d_in[7];
    const float* be   = (const float*)d_in[8];
    const float* Wo   = (const float*)d_in[9];
    const float* bo   = (const float*)d_in[10];
    float* out = (float*)d_out;

    // workspace layout: x[N*D] f32 | agg[N*D] f32 | Wd,We,Wo bf16 [D*D each]
    float* x   = (float*)d_ws;
    float* agg = x + (size_t)N_NODES * DIM;
    __bf16* wd = (__bf16*)(agg + (size_t)N_NODES * DIM);
    __bf16* we = wd + DIM * DIM;
    __bf16* wo = we + DIM * DIM;

    // zero agg + convert weights (deterministic every call)
    prep_kernel<<<(N_NODES * DIM) / 256, 256, 0, stream>>>(Wd, We, Wo, wd, we, wo, agg);

    // x = mish(node @ Wd^T + bd): 1250 row tiles, 16 waves (all col tiles) each
    node_mlp_kernel<<<N_NODES / 16, 512, 0, stream>>>(node, wd, bd, x);

    // fused edge MLP + gather + relu + scatter-add: 20000 row tiles
    edge_mlp_scatter_kernel<<<N_EDGES / 16, 512, 0, stream>>>(
        edge, we, be, x, src, dst, agg);

    // out = mish((x + agg) @ Wo^T + bo)
    out_mlp_kernel<<<N_NODES / 16, 512, 0, stream>>>(x, agg, wo, bo, out);
}
